// S4Layer_6468220747778
// MI455X (gfx1250) — matrix operand order repaired and checked by reading, not compiled, not run on hardware
//
#include <hip/hip_runtime.h>
#include <hip/hip_bf16.h>
#include <math.h>

// ---------------------------------------------------------------------------
// S4D layer, MI455X (gfx1250, wave32).
//   B=8, L=8192, H=256, N=32.
// Bandwidth-bound overall (~300-400MB @ 23.3TB/s). SSM computed as a diagonal
// linear recurrence (4.3 GFLOP) instead of FFT; GLU projection (17.2 GFLOP)
// done with v_wmma_f32_16x16x32_bf16 and fused GLU+residual+LayerNorm epilogue.
// ---------------------------------------------------------------------------

#define B_ 8
#define L_ 8192
#define H_ 256
#define N_ 32

typedef __attribute__((ext_vector_type(16))) __bf16        v16bf;
typedef __attribute__((ext_vector_type(8)))  float         v8f;
typedef __attribute__((ext_vector_type(4)))  unsigned int  u32x4;

union V16U { u32x4 q[2]; v16bf v; };

static __device__ __forceinline__ unsigned short f2bf(float f) {
  unsigned int u = __float_as_uint(f);
  unsigned int r = u + 0x7fffu + ((u >> 16) & 1u);   // round-to-nearest-even
  return (unsigned short)(r >> 16);
}

static __device__ __forceinline__ float wred32(float v) {
#pragma unroll
  for (int o = 16; o > 0; o >>= 1) v += __shfl_xor(v, o, 32);
  return v;
}

// ---------------------------------------------------------------------------
// Kernel 1: discretization params. w = exp(dt*A); Cd = C*(exp(dt*A)-1)/A.
// ---------------------------------------------------------------------------
__global__ void s4_params(const float* __restrict__ log_dt,
                          const float* __restrict__ Arl,
                          const float* __restrict__ Aim,
                          const float* __restrict__ Cre,
                          const float* __restrict__ Cim,
                          float* __restrict__ wr,  float* __restrict__ wi,
                          float* __restrict__ c0r, float* __restrict__ c0i,
                          float* __restrict__ c1r, float* __restrict__ c1i) {
  int idx = blockIdx.x * blockDim.x + threadIdx.x;
  if (idx >= H_ * N_) return;
  int h = idx / N_;
  float dt  = expf(log_dt[h]);
  float Ar  = -expf(Arl[idx]);
  float Ai  = Aim[idx];
  float er  = expf(dt * Ar);
  float s, c;
  sincosf(dt * Ai, &s, &c);
  float wre = er * c, wim = er * s;
  wr[idx] = wre; wi[idx] = wim;
  float den = Ar * Ar + Ai * Ai;
  float gr = wre - 1.0f, gi = wim;
  float Fr = (gr * Ar + gi * Ai) / den;    // (exp(dtA)-1)/A
  float Fi = (gi * Ar - gr * Ai) / den;
  float a0r = Cre[idx],            a0i = Cim[idx];
  float a1r = Cre[H_ * N_ + idx],  a1i = Cim[H_ * N_ + idx];
  c0r[idx] = a0r * Fr - a0i * Fi;
  c0i[idx] = a0r * Fi + a0i * Fr;
  c1r[idx] = a1r * Fr - a1i * Fi;
  c1i[idx] = a1r * Fi + a1i * Fr;
}

// ---------------------------------------------------------------------------
// Kernel 2: W_out (512x256 f32) -> bf16 for WMMA B-fragments.
// ---------------------------------------------------------------------------
__global__ void s4_wconv(const float* __restrict__ w, unsigned short* __restrict__ wbf) {
  int i = blockIdx.x * blockDim.x + threadIdx.x;
  if (i < 2 * H_ * H_) wbf[i] = f2bf(w[i]);
}

// ---------------------------------------------------------------------------
// Kernel 3: transpose x (B,L,H) -> u (B,H,L), 32x32 LDS tiles, coalesced both sides.
// ---------------------------------------------------------------------------
__global__ void __launch_bounds__(256) s4_transpose(const float* __restrict__ x,
                                                    float* __restrict__ u) {
  __shared__ float tile[32][33];
  int blk = blockIdx.x;
  const int NH = H_ / 32;                     // 8
  int ht = blk % NH;
  int lt = (blk / NH) % (L_ / 32);
  int b  = blk / (NH * (L_ / 32));
  int tx = threadIdx.x & 31;
  int ty = threadIdx.x >> 5;                  // 0..7
#pragma unroll
  for (int r = ty; r < 32; r += 8)
    tile[r][tx] = x[((size_t)(b * L_ + lt * 32 + r)) * H_ + ht * 32 + tx];
  __syncthreads();
#pragma unroll
  for (int r = ty; r < 32; r += 8)
    u[((size_t)(b * H_ + ht * 32 + r)) * L_ + lt * 32 + tx] = tile[tx][r];
}

// ---------------------------------------------------------------------------
// Kernel 4: bidirectional diagonal-SSM scan. One wave32 per (b,h); lane = mode.
//   fwd:  S[t] = w*S[t-1] + u[t],  y_f[t] = 2*Re sum_n Cd0*S[t]
//   bwd:  T[t] = w*T[t+1] + u[t+1], y_b[t] = 2*Re sum_n Cd1*T[t]
// Phase 1 writes yssm = y_f + D*u (f32); phase 2 adds y_b, applies exact GELU,
// emits bf16 activations ygelu[b,h,l].
// ---------------------------------------------------------------------------
__global__ void __launch_bounds__(32) s4_scan(const float* __restrict__ u,
                                              const float* __restrict__ wr,
                                              const float* __restrict__ wi,
                                              const float* __restrict__ c0r,
                                              const float* __restrict__ c0i,
                                              const float* __restrict__ c1r,
                                              const float* __restrict__ c1i,
                                              const float* __restrict__ D,
                                              float* __restrict__ yssm,
                                              unsigned short* __restrict__ ygelu) {
  int bh = blockIdx.x;
  int h  = bh % H_;
  int lane = threadIdx.x;
  const float* up = u    + (size_t)bh * L_;
  float*       yp = yssm + (size_t)bh * L_;
  unsigned short* gp = ygelu + (size_t)bh * L_;
  int pi = h * N_ + lane;
  float Wr = wr[pi], Wi = wi[pi];
  float Dh = D[h];

  // ---- forward pass ----
  {
    float Cr = c0r[pi], Ci = c0i[pi];
    float sr = 0.f, si = 0.f;
    for (int t0 = 0; t0 < L_; t0 += 32) {
      float uc = up[t0 + lane];           // coalesced chunk
      float yreg = 0.f;
#pragma unroll
      for (int j = 0; j < 32; ++j) {
        float ut  = __shfl(uc, j, 32);    // broadcast u[t]
        float osr = sr;
        sr = Wr * sr - Wi * si + ut;
        si = Wr * si + Wi * osr;
        float cv = wred32(Cr * sr - Ci * si);
        if (lane == j) yreg = 2.0f * cv;
      }
      yp[t0 + lane] = yreg + Dh * uc;
    }
  }

  // ---- backward pass + GELU + bf16 ----
  {
    float Cr = c1r[pi], Ci = c1i[pi];
    float tr = 0.f, ti = 0.f;
    for (int t0 = L_ - 32; t0 >= 0; t0 -= 32) {
      float uc = up[t0 + lane];
      float yreg = 0.f;
#pragma unroll
      for (int j = 31; j >= 0; --j) {
        float cv = wred32(Cr * tr - Ci * ti);   // uses T[t] (pre-update)
        if (lane == j) yreg = 2.0f * cv;
        float ut  = __shfl(uc, j, 32);
        float otr = tr;
        tr = Wr * tr - Wi * ti + ut;            // -> T[t-1]
        ti = Wr * ti + Wi * otr;
      }
      float v = yp[t0 + lane] + yreg;
      float g = 0.5f * v * (1.0f + erff(v * 0.70710678118654752f));  // exact GELU
      gp[t0 + lane] = f2bf(g);
    }
  }
}

// ---------------------------------------------------------------------------
// Kernel 5: WMMA GEMM (M=B*L, K=256, N=512) + bias + GLU + residual + LayerNorm.
// Block = 256 threads (8 waves) owns one (b, 32-position l-tile): M_tile=32,
// full N=512 -> epilogue has everything needed per position for GLU + LN.
// Wave w: n in [w*64, w*64+64), 2 m-subtiles x 4 n-subtiles accumulators.
// ---------------------------------------------------------------------------
#define APITCH 264                 // bf16 elems per A-row in LDS (pad kills bank conflicts)
#define OPITCH 516                 // f32 elems per out-row in LDS

__global__ void __launch_bounds__(256) s4_gemm(const unsigned short* __restrict__ ygelu,
                                               const unsigned short* __restrict__ wbf,
                                               const float* __restrict__ x,
                                               const float* __restrict__ b_out,
                                               const float* __restrict__ gamma,
                                               const float* __restrict__ beta,
                                               float* __restrict__ out) {
  extern __shared__ char smem[];
  unsigned short (*As)[APITCH] = (unsigned short(*)[APITCH])smem;          // 32 x 264 bf16
  float (*OutS)[OPITCH] = (float(*)[OPITCH])(smem + 32 * APITCH * 2);      // 32 x 516 f32

  int blk = blockIdx.x;
  int b   = blk / (L_ / 32);
  int l0  = (blk % (L_ / 32)) * 32;
  int tid  = threadIdx.x;
  int wave = tid >> 5;
  int lane = tid & 31;

  // ---- stage A tile: ygelu[b, h, l0..l0+31] -> As[l_local][h] (transposed) ----
  {
    int r = tid >> 4;                 // 0..15 : row within 16-row group
    int c = tid & 15;                 // 0..15 : pair of l positions
#pragma unroll
    for (int it = 0; it < 16; ++it) {
      int hh = it * 16 + r;
      unsigned int v = *(const unsigned int*)(ygelu + ((size_t)(b * H_ + hh)) * L_ + l0 + c * 2);
      As[c * 2 + 0][hh] = (unsigned short)(v & 0xffffu);
      As[c * 2 + 1][hh] = (unsigned short)(v >> 16);
    }
  }
  __syncthreads();

  // ---- WMMA main loop over K=256 in steps of 32 ----
  v8f acc[2][4];
#pragma unroll
  for (int mt = 0; mt < 2; ++mt)
#pragma unroll
    for (int nt = 0; nt < 4; ++nt)
#pragma unroll
      for (int r = 0; r < 8; ++r) acc[mt][nt][r] = 0.0f;

  int mrow  = lane & 15;              // A: M row / B: N col / D: N col
  int khalf = lane >> 4;

  for (int k0 = 0; k0 < H_; k0 += 32) {
    // A fragments: elems 0..7 -> K=k0+khalf*8+., elems 8..15 -> K=k0+16+khalf*8+.
    V16U a[2];
#pragma unroll
    for (int mt = 0; mt < 2; ++mt) {
      a[mt].q[0] = *(const u32x4*)&As[mt * 16 + mrow][k0 + khalf * 8];
      a[mt].q[1] = *(const u32x4*)&As[mt * 16 + mrow][k0 + 16 + khalf * 8];
    }
#pragma unroll
    for (int nt = 0; nt < 4; ++nt) {
      // B fragment: lane = column g, elems 0..7 -> K=k0+khalf*8+., elems 8..15 -> K=k0+16+khalf*8+.
      int g = wave * 64 + nt * 16 + mrow;
      const u32x4* wp = (const u32x4*)(wbf + (size_t)g * H_ + k0 + khalf * 8);
      V16U bf; bf.q[0] = wp[0]; bf.q[1] = wp[2];
#pragma unroll
      for (int mt = 0; mt < 2; ++mt)
        acc[mt][nt] = __builtin_amdgcn_wmma_f32_16x16x32_bf16(
            false, a[mt].v, false, bf.v, (short)0, acc[mt][nt], false, false);
    }
  }

  // ---- dump accumulators: OutS[m][n];  D layout: M=(lane/16)*8+r, N=lane%16 ----
#pragma unroll
  for (int mt = 0; mt < 2; ++mt)
#pragma unroll
    for (int nt = 0; nt < 4; ++nt) {
      int n     = wave * 64 + nt * 16 + mrow;
      int mbase = mt * 16 + khalf * 8;
#pragma unroll
      for (int r = 0; r < 8; ++r) OutS[mbase + r][n] = acc[mt][nt][r];
    }
  __syncthreads();

  // ---- epilogue: bias + GLU + residual + LayerNorm; wave handles 4 rows ----
#pragma unroll
  for (int rr = 0; rr < 4; ++rr) {
    int m = wave * 4 + rr;
    int l = l0 + m;
    float vals[8];
    float sum = 0.f, sq = 0.f;
#pragma unroll
    for (int k = 0; k < 8; ++k) {
      int hh = lane + 32 * k;
      float av = OutS[m][hh]       + b_out[hh];
      float gv = OutS[m][hh + H_]  + b_out[hh + H_];
      float z  = av * (1.0f / (1.0f + expf(-gv)));            // GLU
      float rv = z + x[((size_t)(b * L_ + l)) * H_ + hh];      // residual (u = x^T)
      vals[k] = rv; sum += rv; sq += rv * rv;
    }
    sum = wred32(sum); sq = wred32(sq);
    float mu   = sum * (1.0f / (float)H_);
    float var  = sq * (1.0f / (float)H_) - mu * mu;
    float rstd = rsqrtf(var + 1e-5f);
#pragma unroll
    for (int k = 0; k < 8; ++k) {
      int hh = lane + 32 * k;
      out[((size_t)(b * L_ + l)) * H_ + hh] = (vals[k] - mu) * rstd * gamma[hh] + beta[hh];
    }
  }
}

// ---------------------------------------------------------------------------
// Host launcher
// ---------------------------------------------------------------------------
extern "C" void kernel_launch(void* const* d_in, const int* in_sizes, int n_in,
                              void* d_out, int out_size, void* d_ws, size_t ws_size,
                              hipStream_t stream) {
  (void)in_sizes; (void)n_in; (void)out_size; (void)ws_size;
  const float* x      = (const float*)d_in[0];
  const float* log_dt = (const float*)d_in[1];
  const float* Arl    = (const float*)d_in[2];
  const float* Aim    = (const float*)d_in[3];
  const float* Cre    = (const float*)d_in[4];
  const float* Cim    = (const float*)d_in[5];
  const float* D      = (const float*)d_in[6];
  const float* Wout   = (const float*)d_in[7];
  const float* bout   = (const float*)d_in[8];
  const float* gamma  = (const float*)d_in[9];
  const float* beta   = (const float*)d_in[10];

  char* ws = (char*)d_ws;
  size_t off = 0;
  float* u = (float*)(ws + off);              off += (size_t)B_ * H_ * L_ * 4;   // 67 MB
  float* yssm = (float*)(ws + off);           off += (size_t)B_ * H_ * L_ * 4;   // 67 MB
  unsigned short* yg = (unsigned short*)(ws + off); off += (size_t)B_ * H_ * L_ * 2; // 33.5 MB
  float* wr  = (float*)(ws + off);            off += (size_t)H_ * N_ * 4;
  float* wi  = (float*)(ws + off);            off += (size_t)H_ * N_ * 4;
  float* c0r = (float*)(ws + off);            off += (size_t)H_ * N_ * 4;
  float* c0i = (float*)(ws + off);            off += (size_t)H_ * N_ * 4;
  float* c1r = (float*)(ws + off);            off += (size_t)H_ * N_ * 4;
  float* c1i = (float*)(ws + off);            off += (size_t)H_ * N_ * 4;
  unsigned short* wbf = (unsigned short*)(ws + off); off += (size_t)2 * H_ * H_ * 2;

  s4_params<<<(H_ * N_ + 255) / 256, 256, 0, stream>>>(log_dt, Arl, Aim, Cre, Cim,
                                                       wr, wi, c0r, c0i, c1r, c1i);
  s4_wconv<<<(2 * H_ * H_ + 255) / 256, 256, 0, stream>>>(Wout, wbf);
  s4_transpose<<<B_ * (L_ / 32) * (H_ / 32), 256, 0, stream>>>(x, u);
  s4_scan<<<B_ * H_, 32, 0, stream>>>(u, wr, wi, c0r, c0i, c1r, c1i, D, yssm, yg);

  size_t shmem = (size_t)32 * APITCH * 2 + (size_t)32 * OPITCH * 4;   // ~83 KB
  s4_gemm<<<B_ * (L_ / 32), 256, shmem, stream>>>(yg, wbf, x, bout, gamma, beta,
                                                  (float*)d_out);
}
